// SlidingWindowAttention_26697516712503
// MI455X (gfx1250) — matrix operand order repaired and checked by reading, not compiled, not run on hardware
//
#include <hip/hip_runtime.h>
#include <hip/hip_bf16.h>

// ---------------- constants ----------------
#define D_MODEL   1024
#define N_HEADS   16
#define HEAD_DIM  64
#define WINDOW    128
#define N_PERSIST 4
#define KSIZE     3
#define SEQ_L     2048
#define T_REAL    (SEQ_L + N_PERSIST)   // 2052
#define T_PAD     2064                  // 129 * 16
#define M_TILES   (T_PAD / 16)          // 129
#define SCALE     0.125f                // HEAD_DIM^-0.5

typedef __bf16 bf16_t;
typedef __attribute__((ext_vector_type(8)))  __bf16 v8bf;
typedef __attribute__((ext_vector_type(16))) __bf16 v16bf;
typedef __attribute__((ext_vector_type(8)))  float  v8f;

// ---------------- WMMA helpers ----------------
__device__ __forceinline__ v16bf make16(v8bf a, v8bf b) {
    v16bf r;
#pragma unroll
    for (int i = 0; i < 8; ++i) { r[i] = a[i]; r[i + 8] = b[i]; }
    return r;
}

// A fragment (16x32 bf16): lane L (L<16): row M=L, K = k0..k0+15
//                          lane L (L>=16): row M=L-16, K = k0+16..k0+31
__device__ __forceinline__ v16bf load_afrag(const bf16_t* base, int pitch, int m0, int k0) {
    int lane = threadIdx.x & 31;
    const bf16_t* p = base + (size_t)(m0 + (lane & 15)) * pitch + k0 + ((lane >> 4) << 4);
    v8bf lo = *(const v8bf*)p;
    v8bf hi = *(const v8bf*)(p + 8);
    return make16(lo, hi);
}

// B fragment (32x16 bf16) where B[k,n] = src[n][k] (src row-major along k):
// lane L (L<16): col N=L, K = k0..k0+15 ; lane L>=16: col N=L-16, K = k0+16..k0+31
__device__ __forceinline__ v16bf load_bfrag(const bf16_t* base, int pitch, int n0, int k0) {
    int lane = threadIdx.x & 31;
    const bf16_t* p = base + (size_t)(n0 + (lane & 15)) * pitch + k0 + ((lane >> 4) << 4);
    v8bf lo = *(const v8bf*)p;
    v8bf hi = *(const v8bf*)(p + 8);
    return make16(lo, hi);
}

// B fragment where the two 16-wide K halves come from two different row offsets
// (used for P@V where key tiles are non-contiguous: persistent tile + window)
__device__ __forceinline__ v16bf load_bfrag2(const bf16_t* base, int pitch, int n0, int ea, int eb) {
    int lane = threadIdx.x & 31;
    int e0 = (lane < 16) ? ea : eb;
    const bf16_t* p = base + (size_t)(n0 + (lane & 15)) * pitch + e0;
    v8bf lo = *(const v8bf*)p;
    v8bf hi = *(const v8bf*)(p + 8);
    return make16(lo, hi);
}

__device__ __forceinline__ v8f wmma_bf16(v16bf a, v16bf b, v8f c) {
    return __builtin_amdgcn_wmma_f32_16x16x32_bf16(false, a, false, b, (short)0, c, false, false);
}

__device__ __forceinline__ v8f vzero8() {
    v8f z;
#pragma unroll
    for (int i = 0; i < 8; ++i) z[i] = 0.f;
    return z;
}

// C/D layout: lane<16 -> N=lane, M = r (VGPR r); lane>=16 -> N=lane-16, M = 8+r
__device__ __forceinline__ void store_cfrag_f32(float* base, int pitch, int m0, int n0, v8f acc) {
    int lane = threadIdx.x & 31;
    int n  = n0 + (lane & 15);
    int mb = m0 + ((lane >> 4) << 3);
#pragma unroll
    for (int r = 0; r < 8; ++r) base[(size_t)(mb + r) * pitch + n] = acc[r];
}

// ---------------- small utility kernels ----------------
__global__ void k_zero_bf16(bf16_t* p, long n) {
    long i = (long)blockIdx.x * blockDim.x + threadIdx.x;
    if (i < n) p[i] = (bf16_t)0.f;
}

__global__ void k_build_xp(const float* __restrict__ x, const float* __restrict__ pm,
                           bf16_t* __restrict__ xpb) {
    long i = (long)blockIdx.x * blockDim.x + threadIdx.x;
    if (i >= (long)T_PAD * D_MODEL) return;
    int t = (int)(i >> 10), c = (int)(i & 1023);
    float v = 0.f;
    if (t < N_PERSIST)      v = pm[t * D_MODEL + c];
    else if (t < T_REAL)    v = x[(size_t)(t - N_PERSIST) * D_MODEL + c];
    xpb[i] = (bf16_t)v;
}

__global__ void k_cvt_bf16(const float* __restrict__ s, bf16_t* __restrict__ d, long n) {
    long i = (long)blockIdx.x * blockDim.x + threadIdx.x;
    if (i < n) d[i] = (bf16_t)s[i];
}

// ---------------- QKV projection: xp @ W^T (3 matrices) ----------------
// one wave computes a 16x64 strip.  Ping-pong double buffering: two named
// fragment sets (X = k0, Y = k0+32); each set is written only by its loads,
// so no rotation copies are needed.  __launch_bounds__(32,1) lifts the
// per-wave VGPR budget so the ~130 live VGPRs fit without scratch spills.
__global__ void __launch_bounds__(32, 1)
k_gemm_qkv(const bf16_t* __restrict__ xpb,
           const bf16_t* __restrict__ Wq, const bf16_t* __restrict__ Wk,
           const bf16_t* __restrict__ Wv,
           float* __restrict__ Qr, float* __restrict__ Kr, float* __restrict__ Vr) {
    int mt = blockIdx.x;
    int yy = blockIdx.y;            // 0..47
    int which = yy >> 4;            // 0..2
    int nt = yy & 15;               // 0..15 (strips of 64)
    const bf16_t* W = (which == 0) ? Wq : ((which == 1) ? Wk : Wv);
    float* O        = (which == 0) ? Qr : ((which == 1) ? Kr : Vr);
    int m0 = mt * 16, n0 = nt * 64;
    v8f c0 = vzero8(), c1 = vzero8(), c2 = vzero8(), c3 = vzero8();

    v16bf ax  = load_afrag(xpb, D_MODEL, m0, 0);
    v16bf bx0 = load_bfrag(W, D_MODEL, n0 +  0, 0);
    v16bf bx1 = load_bfrag(W, D_MODEL, n0 + 16, 0);
    v16bf bx2 = load_bfrag(W, D_MODEL, n0 + 32, 0);
    v16bf bx3 = load_bfrag(W, D_MODEL, n0 + 48, 0);
    v16bf ay  = load_afrag(xpb, D_MODEL, m0, 32);
    v16bf by0 = load_bfrag(W, D_MODEL, n0 +  0, 32);
    v16bf by1 = load_bfrag(W, D_MODEL, n0 + 16, 32);
    v16bf by2 = load_bfrag(W, D_MODEL, n0 + 32, 32);
    v16bf by3 = load_bfrag(W, D_MODEL, n0 + 48, 32);
#pragma unroll 1
    for (int k0 = 64; k0 < D_MODEL; k0 += 64) {
        c0 = wmma_bf16(ax, bx0, c0);
        c1 = wmma_bf16(ax, bx1, c1);
        c2 = wmma_bf16(ax, bx2, c2);
        c3 = wmma_bf16(ax, bx3, c3);
        ax  = load_afrag(xpb, D_MODEL, m0, k0);
        bx0 = load_bfrag(W, D_MODEL, n0 +  0, k0);
        bx1 = load_bfrag(W, D_MODEL, n0 + 16, k0);
        bx2 = load_bfrag(W, D_MODEL, n0 + 32, k0);
        bx3 = load_bfrag(W, D_MODEL, n0 + 48, k0);
        c0 = wmma_bf16(ay, by0, c0);
        c1 = wmma_bf16(ay, by1, c1);
        c2 = wmma_bf16(ay, by2, c2);
        c3 = wmma_bf16(ay, by3, c3);
        ay  = load_afrag(xpb, D_MODEL, m0, k0 + 32);
        by0 = load_bfrag(W, D_MODEL, n0 +  0, k0 + 32);
        by1 = load_bfrag(W, D_MODEL, n0 + 16, k0 + 32);
        by2 = load_bfrag(W, D_MODEL, n0 + 32, k0 + 32);
        by3 = load_bfrag(W, D_MODEL, n0 + 48, k0 + 32);
    }
    c0 = wmma_bf16(ax, bx0, c0);
    c1 = wmma_bf16(ax, bx1, c1);
    c2 = wmma_bf16(ax, bx2, c2);
    c3 = wmma_bf16(ax, bx3, c3);
    c0 = wmma_bf16(ay, by0, c0);
    c1 = wmma_bf16(ay, by1, c1);
    c2 = wmma_bf16(ay, by2, c2);
    c3 = wmma_bf16(ay, by3, c3);

    store_cfrag_f32(O, D_MODEL, m0, n0 +  0, c0);
    store_cfrag_f32(O, D_MODEL, m0, n0 + 16, c1);
    store_cfrag_f32(O, D_MODEL, m0, n0 + 32, c2);
    store_cfrag_f32(O, D_MODEL, m0, n0 + 48, c3);
}

// ---------------- causal depthwise conv + SiLU (+ optional L2 norm) --------
// one block per token row t; 256 threads, 4 channels each
__global__ void k_conv(const float* __restrict__ Y, const float* __restrict__ w,
                       const float* __restrict__ b, bf16_t* __restrict__ Yb,
                       bf16_t* __restrict__ Vt, int do_norm) {
    int t = blockIdx.x;              // 0..T_REAL-1
    int tid = threadIdx.x;
    __shared__ float red[256];
    float vals[4];
    float ss = 0.f;
#pragma unroll
    for (int i = 0; i < 4; ++i) {
        int c = tid + i * 256;
        float z = b[c];
#pragma unroll
        for (int j = 0; j < KSIZE; ++j) {
            int tt = t - (KSIZE - 1) + j;
            if (tt >= 0) z += w[c * KSIZE + j] * Y[(size_t)tt * D_MODEL + c];
        }
        float s = z / (1.f + __expf(-z));   // SiLU
        vals[i] = s;
        ss += s * s;
    }
    if (do_norm) {
        red[tid] = ss;
        __syncthreads();
        for (int st = 128; st > 0; st >>= 1) {
            if (tid < st) red[tid] += red[tid + st];
            __syncthreads();
        }
        float nrm = sqrtf(red[0]);
        float inv = 1.f / fmaxf(nrm, 1e-12f);
#pragma unroll
        for (int i = 0; i < 4; ++i) vals[i] *= inv;
    }
#pragma unroll
    for (int i = 0; i < 4; ++i) {
        int c = tid + i * 256;
        Yb[(size_t)t * D_MODEL + c] = (bf16_t)vals[i];
        if (Vt) Vt[(size_t)c * T_PAD + t] = (bf16_t)vals[i];   // per-channel transpose
    }
}

// ---------------- windowed attention, one wave per (head, 16-row q tile) ----
// key-tile i (0..nt-1): tile 0 is the persistent tile when the window has
// slid past it, else tiles are just ktlo..qt.  Closed form (no local array).
__device__ __forceinline__ int tile_of(int i, int ktlo) {
    return (ktlo > 0) ? ((i == 0) ? 0 : (ktlo + i - 1)) : i;
}

__global__ void __launch_bounds__(32, 1)
k_attn(const bf16_t* __restrict__ Qb, const bf16_t* __restrict__ Kb,
       const bf16_t* __restrict__ Vt, bf16_t* __restrict__ Ob) {
    __shared__ float  Ssh[16 * 176];
    __shared__ bf16_t Psh[16 * 192];
    __shared__ float  Linv[16];
    int lane = threadIdx.x & 31;
    int qt = blockIdx.x, h = blockIdx.y;
    int q0 = qt * 16;
    int hc = h * HEAD_DIM;

    int lo = q0 - (WINDOW - 1); if (lo < 0) lo = 0;
    int ktlo = lo >> 4;
    int nt = (ktlo > 0) ? (qt - ktlo + 2) : (qt + 1);
    int ncols = nt * 16;

    // Q fragments for this head (K = 64 -> two 32-deep fragments)
    v16bf aq0 = load_afrag(Qb, D_MODEL, q0, hc);
    v16bf aq1 = load_afrag(Qb, D_MODEL, q0, hc + 32);

    // scores: S = Q K^T * scale, masked, into LDS (pipelined over key tiles)
    int n  = lane & 15;
    int mb = (lane >> 4) << 3;
    int kb = tile_of(0, ktlo) * 16;
    v16bf b0 = load_bfrag(Kb, D_MODEL, kb, hc);
    v16bf b1 = load_bfrag(Kb, D_MODEL, kb, hc + 32);
    for (int i = 0; i < nt; ++i) {
        int kbn = (i + 1 < nt) ? tile_of(i + 1, ktlo) * 16 : kb;
        v16bf nb0 = load_bfrag(Kb, D_MODEL, kbn, hc);
        v16bf nb1 = load_bfrag(Kb, D_MODEL, kbn, hc + 32);
        v8f acc = vzero8();
        acc = wmma_bf16(aq0, b0, acc);
        acc = wmma_bf16(aq1, b1, acc);
        int cg = kb + n;
#pragma unroll
        for (int r = 0; r < 8; ++r) {
            int rg = q0 + mb + r;
            bool allowed;
            if (rg < N_PERSIST) {
                allowed = (cg <= rg);
            } else {
                int wlo = rg - (WINDOW - 1); if (wlo < N_PERSIST) wlo = N_PERSIST;
                allowed = (cg < N_PERSIST) || (cg >= wlo && cg <= rg);
            }
            Ssh[(mb + r) * 176 + i * 16 + n] = allowed ? acc[r] * SCALE : -1e30f;
        }
        kb = kbn; b0 = nb0; b1 = nb1;
    }
    __syncthreads();

    // fp32 softmax, probabilities -> bf16 LDS (zero-padded to a multiple of 32)
    int ncp = (ncols + 31) & ~31;
    if (lane < 16) {
        float mx = -1e30f;
        for (int c = 0; c < ncols; ++c) mx = fmaxf(mx, Ssh[lane * 176 + c]);
        float sum = 0.f;
        for (int c = 0; c < ncp; ++c) {
            float e = 0.f;
            if (c < ncols) { e = __expf(Ssh[lane * 176 + c] - mx); sum += e; }
            Psh[lane * 192 + c] = (bf16_t)e;
        }
        Linv[lane] = (sum > 0.f) ? 1.f / sum : 0.f;
    }
    __syncthreads();

    // P @ V via Vt (per-channel transpose). Each WMMA's two 16-K halves map to
    // two (possibly non-adjacent) key tiles -> contiguous 32B loads each.
    // Pipelined: pair j+1's fragments are loaded before pair j's WMMAs.
    int npair = ncp >> 5;
    v8f o0 = vzero8(), o1 = vzero8(), o2 = vzero8(), o3 = vzero8();

    int i0 = 0 < nt - 1 ? 0 : nt - 1;
    int i1 = 1 < nt - 1 ? 1 : nt - 1;
    int ea = tile_of(i0, ktlo) * 16, eb = tile_of(i1, ktlo) * 16;
    v16bf ap  = load_afrag(Psh, 192, 0, 0);
    v16bf bv0 = load_bfrag2(Vt, T_PAD, hc +  0, ea, eb);
    v16bf bv1 = load_bfrag2(Vt, T_PAD, hc + 16, ea, eb);
    v16bf bv2 = load_bfrag2(Vt, T_PAD, hc + 32, ea, eb);
    v16bf bv3 = load_bfrag2(Vt, T_PAD, hc + 48, ea, eb);
    for (int j = 0; j < npair; ++j) {
        int jn = (j + 1 < npair) ? j + 1 : j;
        int ja = 2 * jn     < nt ? 2 * jn     : nt - 1;
        int jb = 2 * jn + 1 < nt ? 2 * jn + 1 : nt - 1;
        int nea = tile_of(ja, ktlo) * 16, neb = tile_of(jb, ktlo) * 16;
        v16bf nap  = load_afrag(Psh, 192, 0, jn * 32);
        v16bf nbv0 = load_bfrag2(Vt, T_PAD, hc +  0, nea, neb);
        v16bf nbv1 = load_bfrag2(Vt, T_PAD, hc + 16, nea, neb);
        v16bf nbv2 = load_bfrag2(Vt, T_PAD, hc + 32, nea, neb);
        v16bf nbv3 = load_bfrag2(Vt, T_PAD, hc + 48, nea, neb);
        o0 = wmma_bf16(ap, bv0, o0);
        o1 = wmma_bf16(ap, bv1, o1);
        o2 = wmma_bf16(ap, bv2, o2);
        o3 = wmma_bf16(ap, bv3, o3);
        ap = nap; bv0 = nbv0; bv1 = nbv1; bv2 = nbv2; bv3 = nbv3;
    }

    // scale rows by 1/l and store (skip padded query rows; pads pre-zeroed)
#pragma unroll
    for (int r = 0; r < 8; ++r) {
        int t = q0 + mb + r;
        if (t < T_REAL) {
            float li = Linv[mb + r];
            Ob[(size_t)t * D_MODEL + hc +  0 + n] = (bf16_t)(o0[r] * li);
            Ob[(size_t)t * D_MODEL + hc + 16 + n] = (bf16_t)(o1[r] * li);
            Ob[(size_t)t * D_MODEL + hc + 32 + n] = (bf16_t)(o2[r] * li);
            Ob[(size_t)t * D_MODEL + hc + 48 + n] = (bf16_t)(o3[r] * li);
        }
    }
}

// ---------------- output projection: O @ Wo^T, rows 4..2051 -> d_out -------
// one wave computes a 16x64 strip; ping-pong double buffered like k_gemm_qkv
__global__ void __launch_bounds__(32, 1)
k_oproj(const bf16_t* __restrict__ Ob, const bf16_t* __restrict__ Wob,
        float* __restrict__ out) {
    int m0 = blockIdx.x * 16, n0 = blockIdx.y * 64;
    v8f c0 = vzero8(), c1 = vzero8(), c2 = vzero8(), c3 = vzero8();

    v16bf ax  = load_afrag(Ob, D_MODEL, m0, 0);
    v16bf bx0 = load_bfrag(Wob, D_MODEL, n0 +  0, 0);
    v16bf bx1 = load_bfrag(Wob, D_MODEL, n0 + 16, 0);
    v16bf bx2 = load_bfrag(Wob, D_MODEL, n0 + 32, 0);
    v16bf bx3 = load_bfrag(Wob, D_MODEL, n0 + 48, 0);
    v16bf ay  = load_afrag(Ob, D_MODEL, m0, 32);
    v16bf by0 = load_bfrag(Wob, D_MODEL, n0 +  0, 32);
    v16bf by1 = load_bfrag(Wob, D_MODEL, n0 + 16, 32);
    v16bf by2 = load_bfrag(Wob, D_MODEL, n0 + 32, 32);
    v16bf by3 = load_bfrag(Wob, D_MODEL, n0 + 48, 32);
#pragma unroll 1
    for (int k0 = 64; k0 < D_MODEL; k0 += 64) {
        c0 = wmma_bf16(ax, bx0, c0);
        c1 = wmma_bf16(ax, bx1, c1);
        c2 = wmma_bf16(ax, bx2, c2);
        c3 = wmma_bf16(ax, bx3, c3);
        ax  = load_afrag(Ob, D_MODEL, m0, k0);
        bx0 = load_bfrag(Wob, D_MODEL, n0 +  0, k0);
        bx1 = load_bfrag(Wob, D_MODEL, n0 + 16, k0);
        bx2 = load_bfrag(Wob, D_MODEL, n0 + 32, k0);
        bx3 = load_bfrag(Wob, D_MODEL, n0 + 48, k0);
        c0 = wmma_bf16(ay, by0, c0);
        c1 = wmma_bf16(ay, by1, c1);
        c2 = wmma_bf16(ay, by2, c2);
        c3 = wmma_bf16(ay, by3, c3);
        ay  = load_afrag(Ob, D_MODEL, m0, k0 + 32);
        by0 = load_bfrag(Wob, D_MODEL, n0 +  0, k0 + 32);
        by1 = load_bfrag(Wob, D_MODEL, n0 + 16, k0 + 32);
        by2 = load_bfrag(Wob, D_MODEL, n0 + 32, k0 + 32);
        by3 = load_bfrag(Wob, D_MODEL, n0 + 48, k0 + 32);
    }
    c0 = wmma_bf16(ax, bx0, c0);
    c1 = wmma_bf16(ax, bx1, c1);
    c2 = wmma_bf16(ax, bx2, c2);
    c3 = wmma_bf16(ax, bx3, c3);
    c0 = wmma_bf16(ay, by0, c0);
    c1 = wmma_bf16(ay, by1, c1);
    c2 = wmma_bf16(ay, by2, c2);
    c3 = wmma_bf16(ay, by3, c3);

    int lane = threadIdx.x & 31;
    int n  = lane & 15;
    int mb = m0 + ((lane >> 4) << 3);
#pragma unroll
    for (int r = 0; r < 8; ++r) {
        int t = mb + r;
        if (t >= N_PERSIST && t < T_REAL) {
            float* orow = out + (size_t)(t - N_PERSIST) * D_MODEL;
            orow[n0 +  0 + n] = c0[r];
            orow[n0 + 16 + n] = c1[r];
            orow[n0 + 32 + n] = c2[r];
            orow[n0 + 48 + n] = c3[r];
        }
    }
}

// ---------------- launcher ----------------
extern "C" void kernel_launch(void* const* d_in, const int* in_sizes, int n_in,
                              void* d_out, int out_size, void* d_ws, size_t ws_size,
                              hipStream_t stream) {
    const float* x  = (const float*)d_in[0];
    const float* pm = (const float*)d_in[1];
    const float* Wq = (const float*)d_in[2];
    const float* Wk = (const float*)d_in[3];
    const float* Wv = (const float*)d_in[4];
    const float* Wo = (const float*)d_in[5];
    const float* qw = (const float*)d_in[6];
    const float* qbv = (const float*)d_in[7];
    const float* kw = (const float*)d_in[8];
    const float* kbv = (const float*)d_in[9];
    const float* vw = (const float*)d_in[10];
    const float* vbv = (const float*)d_in[11];
    float* out = (float*)d_out;

    char* base = (char*)d_ws;
    size_t off = 0;
    auto take = [&](size_t bytes) -> char* {
        char* p = base + off;
        off += (bytes + 255) & ~(size_t)255;
        return p;
    };
    // contiguous bf16 block (zeroed each launch): Qb | Kb | Vb | Vt | Ob
    bf16_t* Qbf = (bf16_t*)take((size_t)T_PAD * D_MODEL * 2);
    bf16_t* Kbf = (bf16_t*)take((size_t)T_PAD * D_MODEL * 2);
    bf16_t* Vbf = (bf16_t*)take((size_t)T_PAD * D_MODEL * 2);
    bf16_t* Vt  = (bf16_t*)take((size_t)D_MODEL * T_PAD * 2);
    bf16_t* Ob  = (bf16_t*)take((size_t)T_PAD * D_MODEL * 2);
    bf16_t* xpb = (bf16_t*)take((size_t)T_PAD * D_MODEL * 2);
    bf16_t* Wqb = (bf16_t*)take((size_t)D_MODEL * D_MODEL * 2);
    bf16_t* Wkb = (bf16_t*)take((size_t)D_MODEL * D_MODEL * 2);
    bf16_t* Wvb = (bf16_t*)take((size_t)D_MODEL * D_MODEL * 2);
    bf16_t* Wob = (bf16_t*)take((size_t)D_MODEL * D_MODEL * 2);
    float*  Qr  = (float*)take((size_t)T_PAD * D_MODEL * 4);
    float*  Kr  = (float*)take((size_t)T_PAD * D_MODEL * 4);
    float*  Vr  = (float*)take((size_t)T_PAD * D_MODEL * 4);

    long zn = (long)T_PAD * D_MODEL * 5;   // Qbf..Ob contiguous
    k_zero_bf16<<<dim3((unsigned)((zn + 255) / 256)), 256, 0, stream>>>(Qbf, zn);

    long xn = (long)T_PAD * D_MODEL;
    k_build_xp<<<dim3((unsigned)((xn + 255) / 256)), 256, 0, stream>>>(x, pm, xpb);

    long wn = (long)D_MODEL * D_MODEL;
    unsigned wb = (unsigned)((wn + 255) / 256);
    k_cvt_bf16<<<dim3(wb), 256, 0, stream>>>(Wq, Wqb, wn);
    k_cvt_bf16<<<dim3(wb), 256, 0, stream>>>(Wk, Wkb, wn);
    k_cvt_bf16<<<dim3(wb), 256, 0, stream>>>(Wv, Wvb, wn);
    k_cvt_bf16<<<dim3(wb), 256, 0, stream>>>(Wo, Wob, wn);

    // 48 = 3 matrices x 16 strips of 64 columns
    k_gemm_qkv<<<dim3(M_TILES, 48), 32, 0, stream>>>(xpb, Wqb, Wkb, Wvb, Qr, Kr, Vr);

    k_conv<<<dim3(T_REAL), 256, 0, stream>>>(Qr, qw, qbv, Qbf, nullptr, 1);
    k_conv<<<dim3(T_REAL), 256, 0, stream>>>(Kr, kw, kbv, Kbf, nullptr, 1);
    k_conv<<<dim3(T_REAL), 256, 0, stream>>>(Vr, vw, vbv, Vbf, Vt, 0);

    k_attn<<<dim3(M_TILES, N_HEADS), 32, 0, stream>>>(Qbf, Kbf, Vt, Ob);

    k_oproj<<<dim3(M_TILES, D_MODEL / 64), 32, 0, stream>>>(Ob, Wob, out);
}
